// SimpleRNN_37177236914792
// MI455X (gfx1250) — matrix operand order repaired and checked by reading, not compiled, not run on hardware
//
#include <hip/hip_runtime.h>

typedef __attribute__((ext_vector_type(16))) _Float16 v16h;
typedef __attribute__((ext_vector_type(8)))  float    v8f;

#define T_STEPS 512
#define BATCH   1024
#define INF     128
#define HID     256
#define OUTC    4
#define KTOT    (INF + HID)   // 384 contraction length (combined = [x | h])
#define KP      (KTOT / 2)    // 192 f16-pair rows of B
#define NB      288           // 256 hidden + 4 logit + 28 zero cols (9 waves x 2 tiles)
#define NBP     292           // padded u32 row stride: 4*292 % 64 == 16 -> halves on disjoint banks
#define CSTR    198           // padded u32 pair stride for combined rows
#define ROWS    16            // batch rows per workgroup
#define NWAVES  9
#define NTHR    (NWAVES * 32)
#define KB_N    (KTOT / 32)   // 12 K-iterations

union Frag { v16h v; uint32_t u[8]; };

__global__ __launch_bounds__(NTHR, 1)
void rnn_fused_wmma(const float* __restrict__ inp, const float* __restrict__ hidden,
                    const float* __restrict__ W_i2h, const float* __restrict__ b_i2h,
                    const float* __restrict__ W_i2o, const float* __restrict__ b_i2o,
                    const float* __restrict__ actions, float* __restrict__ out)
{
    // B matrix resident in LDS for all 512 steps, pre-packed as f16 pairs:
    // Bp[p*NBP + n] = { B[2p][n], B[2p+1][n] }, B[k][n] = W_i2h[n][k] (n<256),
    // W_i2o[n-256][k] (256<=n<260), else 0.
    __shared__ uint32_t Bp[KP * NBP];            // ~224 KB
    __shared__ uint32_t combp[2 * ROWS * CSTR];  // double-buffered [x_t | h] f16 pairs
    __shared__ float    act_s[4 * HID];
    __shared__ float    bias_s[NB];
    __shared__ float    outbuf[ROWS * OUTC];
    __shared__ int      q_s[ROWS];

    const int tid  = threadIdx.x;
    const int w    = tid >> 5;        // wave id 0..8
    const int lane = tid & 31;
    const int half = lane >> 4;
    const int lq   = lane & 15;
    const int b0   = blockIdx.x * ROWS;
    const int wu   = __builtin_amdgcn_readfirstlane(w);  // provably wave-uniform role id

    float* hs0 = out + (size_t)T_STEPS * BATCH * OUTC;   // hs[0] output region

    // ---------------- one-time preload ----------------
    for (int idx = tid; idx < KP * NB; idx += NTHR) {
        int p = idx / NB, n = idx - p * NB;
        float lo = 0.f, hi = 0.f;
        if (n < HID)             { lo = W_i2h[n * KTOT + 2 * p]; hi = W_i2h[n * KTOT + 2 * p + 1]; }
        else if (n < HID + OUTC) { int o = n - HID;
                                   lo = W_i2o[o * KTOT + 2 * p]; hi = W_i2o[o * KTOT + 2 * p + 1]; }
        union { _Float16 h[2]; uint32_t u; } pk;
        pk.h[0] = (_Float16)lo; pk.h[1] = (_Float16)hi;
        Bp[p * NBP + n] = pk.u;
    }
    for (int idx = tid; idx < 4 * HID; idx += NTHR) act_s[idx] = actions[idx];
    for (int idx = tid; idx < NB; idx += NTHR) {
        float b = 0.f;
        if (idx < HID) b = b_i2h[idx];
        else if (idx < HID + OUTC) b = b_i2o[idx - HID];
        bias_s[idx] = b;
    }
    // initial hidden state -> comb buffer 0, h-part (pairs 64..191)
    for (int idx = tid; idx < ROWS * (HID / 2); idx += NTHR) {
        int m = idx >> 7, pj = idx & 127;
        const float2 hv = *(const float2*)&hidden[(size_t)(b0 + m) * HID + 2 * pj];
        union { _Float16 h[2]; uint32_t u; } pk;
        pk.h[0] = (_Float16)hv.x; pk.h[1] = (_Float16)hv.y;
        combp[(0 * ROWS + m) * CSTR + (INF / 2) + pj] = pk.u;
    }
    __syncthreads();

    // per-lane column ids + bias registers (uniform across the step loop)
    const int n0 = 32 * w + lq, n1 = n0 + 16;
    const float bh0 = bias_s[n0];
    const float bh1 = bias_s[n1];

    int cur = 0;
    for (int t = 0; t < T_STEPS; ++t) {
        // ---- stage x_t tile (f32 -> f16 pairs) into comb[cur] pairs 0..63 ----
        const float* xrow = inp + (size_t)t * BATCH * INF + (size_t)b0 * INF;
        for (int idx = tid; idx < ROWS * (INF / 2); idx += NTHR) {
            int m = idx >> 6, pj = idx & 63;
            const float2 xv = *(const float2*)&xrow[(size_t)m * INF + 2 * pj];
            union { _Float16 h[2]; uint32_t u; } pk;
            pk.h[0] = (_Float16)xv.x; pk.h[1] = (_Float16)xv.y;
            combp[(cur * ROWS + m) * CSTR + pj] = pk.u;
        }
        __syncthreads();  // B1: combined[t] fully staged

        // prefetch next step's x tile while WMMAs run
        if (t + 1 < T_STEPS && tid < 128) {
            const char* nx = (const char*)(inp + (size_t)(t + 1) * BATCH * INF + (size_t)b0 * INF);
            __builtin_prefetch(nx + tid * 64, 0, 0);
        }

        // ---- branch-free, software-pipelined GEMM: every wave does 2 col-tiles ----
        Frag af[2], b0f[2], b1f[2];
        v8f c0, c1;
        #pragma unroll
        for (int i = 0; i < 8; ++i) { c0[i] = bh0; c1[i] = bh1; }

        const uint32_t* crow  = &combp[(cur * ROWS + lq) * CSTR];
        const uint32_t* bcol0 = &Bp[n0];
        const uint32_t* bcol1 = &Bp[n1];

#define LOAD_FRAGS(buf, kb)                                                        \
        {                                                                          \
            const int pb_ = (kb) * 16;                                             \
            _Pragma("unroll")                                                      \
            for (int i = 0; i < 4; ++i) af[buf].u[i]     = crow[pb_ + 4 * half + i];      \
            _Pragma("unroll")                                                      \
            for (int i = 0; i < 4; ++i) af[buf].u[4 + i] = crow[pb_ + 8 + 4 * half + i];  \
            _Pragma("unroll")                                                      \
            for (int i = 0; i < 8; ++i) b0f[buf].u[i] = bcol0[(pb_ + 8 * (i >> 2) + 4 * half + (i & 3)) * NBP]; \
            _Pragma("unroll")                                                      \
            for (int i = 0; i < 8; ++i) b1f[buf].u[i] = bcol1[(pb_ + 8 * (i >> 2) + 4 * half + (i & 3)) * NBP]; \
        }

        LOAD_FRAGS(0, 0)
        #pragma unroll
        for (int kb = 0; kb < KB_N; ++kb) {
            const int bc = kb & 1, bn = bc ^ 1;
            if (kb + 1 < KB_N) LOAD_FRAGS(bn, kb + 1)   // loads for kb+1 in flight over kb's WMMAs
            c0 = __builtin_amdgcn_wmma_f32_16x16x32_f16(false, af[bc].v, false, b0f[bc].v, (short)0, c0, false, false);
            c1 = __builtin_amdgcn_wmma_f32_16x16x32_f16(false, af[bc].v, false, b1f[bc].v, (short)0, c1, false, false);
        }
#undef LOAD_FRAGS

        float th0[8], th1[8];
        if (wu < 8) {
            // tanh (pre-gating new_h); q not needed yet
            #pragma unroll
            for (int i = 0; i < 8; ++i) { th0[i] = tanhf(c0[i]); th1[i] = tanhf(c1[i]); }
            if (t == 0) {  // hs[0] output = new_h at t=0, before gating
                #pragma unroll
                for (int i = 0; i < 8; ++i) {
                    int m = i + 8 * half;
                    hs0[(size_t)(b0 + m) * HID + n0] = th0[i];
                    hs0[(size_t)(b0 + m) * HID + n1] = th1[i];
                }
            }
        } else {
            // logits live in c0 columns n0=256+lq; stream to global + LDS, in-wave argmax
            if (lq < OUTC) {
                #pragma unroll
                for (int i = 0; i < 8; ++i) {
                    int m = i + 8 * half;
                    float v = c0[i];
                    out[((size_t)t * BATCH + b0 + m) * OUTC + lq] = v;
                    outbuf[m * OUTC + lq] = v;
                }
            }
            __asm__ volatile("s_wait_dscnt 0" ::: "memory");  // same-wave LDS RAW
            if (lq == 0) {
                #pragma unroll
                for (int i = 0; i < 8; ++i) {
                    int m = i + 8 * half;
                    float best = outbuf[m * OUTC + 0]; int q = 0;
                    #pragma unroll
                    for (int n = 1; n < OUTC; ++n) {
                        float v = outbuf[m * OUTC + n];
                        if (v > best) { best = v; q = n; }   // strict > keeps first max (jnp.argmax)
                    }
                    q_s[m] = q;
                }
            }
        }

        __syncthreads();  // B2: q_s ready; all comb[cur] reads complete

        // ---- gate and write h_t (f16) into comb[nxt] h-part ----
        const int nxt = cur ^ 1;
        if (wu < 8) {
            _Float16* ch = (_Float16*)combp;
            #pragma unroll
            for (int i = 0; i < 8; ++i) {
                int m = i + 8 * half;
                int q = q_s[m];
                float g0 = th0[i] * (1.f + act_s[q * HID + n0]);
                float g1 = th1[i] * (1.f + act_s[q * HID + n1]);
                ch[((nxt * ROWS + m) * CSTR + (INF / 2) + (n0 >> 1)) * 2 + (n0 & 1)] = (_Float16)g0;
                ch[((nxt * ROWS + m) * CSTR + (INF / 2) + (n1 >> 1)) * 2 + (n1 & 1)] = (_Float16)g1;
            }
        }
        cur = nxt;
    }
}

extern "C" void kernel_launch(void* const* d_in, const int* in_sizes, int n_in,
                              void* d_out, int out_size, void* d_ws, size_t ws_size,
                              hipStream_t stream) {
    (void)in_sizes; (void)n_in; (void)d_ws; (void)ws_size; (void)out_size;
    const float* inp     = (const float*)d_in[0];
    const float* hidden  = (const float*)d_in[1];
    const float* W_i2h   = (const float*)d_in[2];
    const float* b_i2h   = (const float*)d_in[3];
    const float* W_i2o   = (const float*)d_in[4];
    const float* b_i2o   = (const float*)d_in[5];
    const float* actions = (const float*)d_in[6];
    float* out = (float*)d_out;

    rnn_fused_wmma<<<BATCH / ROWS, NTHR, 0, stream>>>(
        inp, hidden, W_i2h, b_i2h, W_i2o, b_i2o, actions, out);
}
